// PCT_Encoder_32126355374754
// MI455X (gfx1250) — matrix operand order repaired and checked by reading, not compiled, not run on hardware
//
#include <hip/hip_runtime.h>
#include <math.h>

// ---------------------------------------------------------------------------
// PCT encoder for MI455X (gfx1250, wave32, WMMA).
// All channel matmuls -> v_wmma_f32_16x16x32_f16 (f16 in, f32 acc).
// A operands pre-packed into WMMA register order (contiguous v16h per lane).
// B operands staged transposed in LDS -> two aligned ds_load_b128 per lane.
// Irregular stages (FPS / kNN / gather / softmax / BN stats) -> VALU kernels.
// ---------------------------------------------------------------------------

typedef __attribute__((ext_vector_type(16))) _Float16 v16h;
typedef __attribute__((ext_vector_type(8)))  _Float16 v8h;
typedef __attribute__((ext_vector_type(8)))  float    v8f;

#define BATCH 32
#define LDS_STRIDE 40   // padded col stride (elems): 80B -> 16B-aligned, no bank conflicts

// ---------------------------------------------------------------------------
// Pack an A operand into WMMA register order:
//   dst[(((b*(O/16)+ot)*(C/32)+ch)*32 + lane)*16 + e] =
//       src[b*aBatch + o*aRow + c*aCol],
//   o = ot*16 + (lane&15),  c = ch*32 + Ka(e, lane>>4)
// Works for f32 weights (cvt fused) and per-batch f16 matrices.
// ---------------------------------------------------------------------------
template <typename SrcT>
__global__ void packA_kernel(const SrcT* __restrict__ src, size_t aBatch,
                             int aRow, int aCol, _Float16* __restrict__ dst,
                             int O, int C, int total) {
  int i = blockIdx.x * 256 + threadIdx.x;
  if (i >= total) return;
  int e = i & 15, lane = (i >> 4) & 31;
  int rest = i >> 9;
  int nch = C >> 5;
  int ch = rest % nch; rest /= nch;
  int ot = rest % (O >> 4);
  int b  = rest / (O >> 4);
  int half = lane >> 4, mrow = lane & 15;
  int j = e >> 1, r = e & 1;
  int Ka = (half << 4) + e;
  int o = (ot << 4) + mrow, c = (ch << 5) + Ka;
  dst[i] = (_Float16)src[(size_t)b * aBatch + (size_t)o * aRow + (size_t)c * aCol];
}

// ---------------------------------------------------------------------------
// Generic WMMA GEMM:  Y[b][Yoff+o][m] = sum_c A(b,o,c) * X[b][Xcoff+c][m] (+bias[o])
// Apk: packed A (aPB = per-batch element stride, 0 for shared weights).
// Block = 256 threads = 8 waves -> 32(o) x 64(m) super-tile.
// NOTE: grids are launched exact (O%32==0, M%64==0) - no partial tiles.
// ---------------------------------------------------------------------------
template <typename OutT>
__global__ void __launch_bounds__(256) gemm16_kernel(
    const _Float16* __restrict__ Apk, size_t aPB,
    const _Float16* __restrict__ X, int XCB, int Xcoff,
    OutT* __restrict__ Y, int YOB, int Yoff,
    const float* __restrict__ bias,
    int M, int C, int O)
{
  const int tid  = threadIdx.x;
  const int lane = tid & 31;
  const int w    = tid >> 5;
  const int half = lane >> 4;
  const int mn   = lane & 15;
  const int o0   = blockIdx.y * 32 + ((w >> 2) << 4);
  const int mblk = blockIdx.x * 64;
  const int m0   = mblk + ((w & 3) << 4);
  const int colB = ((w & 3) << 4) + mn;   // lane's column within the 64-wide tile
  const int b    = blockIdx.z;
  const int nch  = C >> 5;

  const size_t abase = (size_t)b * aPB + ((size_t)(o0 >> 4) * nch << 9) + (lane << 4);
  const size_t xbase = ((size_t)b * XCB + Xcoff) * (size_t)M;

  __shared__ _Float16 xs[64 * LDS_STRIDE];   // transposed tile: [col][K]

  const int crow = tid >> 3;              // 0..31 : K row this thread copies
  const int ccol = (tid & 7) << 3;        // 0..56 : first of 8 cols it copies

  v8f acc = {};
  for (int c0 = 0; c0 < C; c0 += 32) {
    // ---- coalesced 128-bit global read, transposed scatter into LDS ----
    {
      const _Float16* gp = X + xbase + (size_t)(c0 + crow) * M + mblk + ccol;
      uint4 t4 = *(const uint4*)gp;
      const _Float16* tp = (const _Float16*)&t4;
#pragma unroll
      for (int u = 0; u < 8; ++u) xs[(ccol + u) * LDS_STRIDE + crow] = tp[u];
      if (c0 + 32 < C)  // speculative prefetch of next K-chunk
        __builtin_prefetch(gp + (size_t)32 * M, 0, 1);
    }
    __syncthreads();

    // A: one contiguous packed v16h per lane (2x global_load_b128)
    const v16h a = *(const v16h*)(Apk + abase + ((size_t)(c0 >> 5) << 9));
    // B: lane's 16 contiguous K values at its column (2x ds_load_b128)
    const _Float16* bp = &xs[colB * LDS_STRIDE + (half << 4)];
    v8h b0 = *(const v8h*)bp;
    v8h b1 = *(const v8h*)(bp + 8);
    v16h bb;
#pragma unroll
    for (int e = 0; e < 8; ++e) { bb[e] = b0[e]; bb[e + 8] = b1[e]; }

    acc = __builtin_amdgcn_wmma_f32_16x16x32_f16(false, a, false, bb,
                                                 (short)0, acc, false, false);
    __syncthreads();
  }
#pragma unroll
  for (int j = 0; j < 8; ++j) {
    const int orow = o0 + (half << 3) + j;      // C/D: M = j + 8*half
    float vv = acc[j];
    if (bias) vv += bias[orow];
    Y[((size_t)b * YOB + Yoff + orow) * (size_t)M + m0 + mn] = (OutT)vv;
  }
}

// ---------------------------------------------------------------------------
// Gather-fused WMMA GEMM for local_op conv1:
//   feat(b, c, s, k) = c <  C0 ? SRC[b][c][knn(s,k)] - SRC[b][c][fps(s)]
//                    : SRC[b][c-C0][fps(s)]
//   Y[b][o][s*32+k] = sum_c Wpk(o,c) * feat
// c<C0 split is chunk-uniform (C0 % 32 == 0) -> scalar branch. Differenced
// feature tile staged transposed in LDS; per-column indices hoisted to LDS.
// ---------------------------------------------------------------------------
__global__ void __launch_bounds__(256) gemm16_gather_kernel(
    const _Float16* __restrict__ Wpk,           // packed O x C weights
    const _Float16* __restrict__ SRC, int C0, int Nsrc,
    const int* __restrict__ knn, const int* __restrict__ fps, int S,
    _Float16* __restrict__ Y, int M, int C, int O)
{
  const int tid  = threadIdx.x;
  const int lane = tid & 31;
  const int w    = tid >> 5;
  const int half = lane >> 4, mn = lane & 15;
  const int o0   = blockIdx.y * 32 + ((w >> 2) << 4);
  const int mblk = blockIdx.x * 64;
  const int m0   = mblk + ((w & 3) << 4);
  const int colB = ((w & 3) << 4) + mn;
  const int b    = blockIdx.z;
  const int nch  = C >> 5;

  __shared__ _Float16 xs[64 * LDS_STRIDE];
  __shared__ int sidx[64], sfidx[64];

  if (tid < 64) {
    const int m = mblk + tid, s = m >> 5, k = m & 31;
    sidx[tid]  = knn[((size_t)b * S + s) * 32 + k];
    sfidx[tid] = fps[(size_t)b * S + s];
  }
  __syncthreads();

  const _Float16* srcb = SRC + (size_t)b * C0 * Nsrc;
  const size_t abase = ((size_t)(o0 >> 4) * nch << 9) + (lane << 4);
  const int crow = tid >> 3;
  const int ccol = (tid & 7) << 3;

  v8f acc = {};
  for (int c0 = 0; c0 < C; c0 += 32) {
    // ---- stage differenced/center feature tile (chunk-uniform branch) ----
    if (c0 < C0) {
      const _Float16* rowp = srcb + (size_t)(c0 + crow) * Nsrc;
#pragma unroll
      for (int u = 0; u < 8; ++u) {
        const int col = ccol + u;
        xs[col * LDS_STRIDE + crow] =
            (_Float16)(rowp[sidx[col]] - rowp[sfidx[col]]);
      }
    } else {
      const _Float16* rowp = srcb + (size_t)(c0 - C0 + crow) * Nsrc;
#pragma unroll
      for (int u = 0; u < 8; ++u) {
        const int col = ccol + u;
        xs[col * LDS_STRIDE + crow] = rowp[sfidx[col]];
      }
    }
    __syncthreads();

    const v16h a = *(const v16h*)(Wpk + abase + ((size_t)(c0 >> 5) << 9));
    const _Float16* bp = &xs[colB * LDS_STRIDE + (half << 4)];
    v8h b0 = *(const v8h*)bp;
    v8h b1 = *(const v8h*)(bp + 8);
    v16h bb;
#pragma unroll
    for (int e = 0; e < 8; ++e) { bb[e] = b0[e]; bb[e + 8] = b1[e]; }

    acc = __builtin_amdgcn_wmma_f32_16x16x32_f16(false, a, false, bb,
                                                 (short)0, acc, false, false);
    __syncthreads();
  }
#pragma unroll
  for (int j = 0; j < 8; ++j) {
    const int orow = o0 + (half << 3) + j;
    Y[((size_t)b * O + orow) * (size_t)M + m0 + mn] = (_Float16)acc[j];
  }
}

// ----------------------------- small kernels -------------------------------

__global__ void cvt_f32_f16_kernel(const float* __restrict__ s,
                                   _Float16* __restrict__ d, int n) {
  int i = blockIdx.x * 256 + threadIdx.x;
  if (i < n) d[i] = (_Float16)s[i];
}

__global__ void zero_kernel(float* p, int n) {
  int i = blockIdx.x * 256 + threadIdx.x;
  if (i < n) p[i] = 0.f;
}

// conv1: C=3, too thin for WMMA -> VALU.
__global__ void conv1_kernel(const float* __restrict__ x,   // (B,2048,3)
                             const float* __restrict__ Wc,  // (64,3)
                             _Float16* __restrict__ out) {  // (B,64,2048)
  int i = blockIdx.x * 256 + threadIdx.x;
  if (i >= BATCH * 64 * 2048) return;
  int n = i & 2047, o = (i >> 11) & 63, b = i >> 17;
  const float* p = &x[((size_t)b * 2048 + n) * 3];
  float v = Wc[o * 3 + 0] * p[0] + Wc[o * 3 + 1] * p[1] + Wc[o * 3 + 2] * p[2];
  out[((size_t)b * 64 + o) * 2048 + n] = (_Float16)v;
}

// BN stats: per-(channel,batch) block reduces M elements, atomics into sums.
__global__ void bn_stats_kernel(const _Float16* __restrict__ src, int SCB,
                                int soff, int M, float* __restrict__ sums,
                                float* __restrict__ sumsq) {
  int c = blockIdx.x, b = blockIdx.y, t = threadIdx.x;
  const _Float16* p = src + ((size_t)b * SCB + soff + c) * (size_t)M;
  float s = 0.f, q = 0.f;
  for (int m = t; m < M; m += 256) { float v = (float)p[m]; s += v; q += v * v; }
  __shared__ float ss[256], sq[256];
  ss[t] = s; sq[t] = q; __syncthreads();
  for (int st = 128; st > 0; st >>= 1) {
    if (t < st) { ss[t] += ss[t + st]; sq[t] += sq[t + st]; }
    __syncthreads();
  }
  if (t == 0) { atomicAdd(&sums[c], ss[0]); atomicAdd(&sumsq[c], sq[0]); }
}

// BN apply + activation (+ optional residual add AFTER activation).
// mode 0: relu, mode 1: leaky 0.2
__global__ void bn_apply_kernel(const _Float16* __restrict__ src, int SCB, int soff,
                                _Float16* __restrict__ dst, int DCB, int doff,
                                const _Float16* __restrict__ res, int RCB, int roff,
                                const float* __restrict__ g, const float* __restrict__ bt,
                                const float* __restrict__ sums, const float* __restrict__ sumsq,
                                float invCnt, int C, int M, int total, int mode) {
  int i = blockIdx.x * 256 + threadIdx.x;
  if (i >= total) return;
  int m = i % M, c = (i / M) % C, b = i / (M * C);
  float mean = sums[c] * invCnt;
  float var  = sumsq[c] * invCnt - mean * mean;
  float rs   = rsqrtf(var + 1e-5f);
  float x = (float)src[((size_t)b * SCB + soff + c) * M + m];
  float y = g[c] * (x - mean) * rs + bt[c];
  y = (mode == 1) ? (y > 0.f ? y : 0.2f * y) : fmaxf(y, 0.f);
  if (res) y += (float)res[((size_t)b * RCB + roff + c) * M + m];
  dst[((size_t)b * DCB + doff + c) * M + m] = (_Float16)y;
}

// Farthest point sampling: one block per batch, serial npoint iterations.
__global__ void fps_kernel(const float* __restrict__ xyz, int N, int np,
                           int* __restrict__ out) {
  int b = blockIdx.x, t = threadIdx.x;
  __shared__ float dist[2048];
  __shared__ float rv[256];
  __shared__ int   ri[256];
  __shared__ float cent[3];
  __shared__ int   sfar;
  for (int n = t; n < N; n += 256) dist[n] = 1e10f;
  int far = 0;
  for (int i = 0; i < np; ++i) {
    if (t == 0) {
      out[b * np + i] = far;
      const float* p = &xyz[((size_t)b * N + far) * 3];
      cent[0] = p[0]; cent[1] = p[1]; cent[2] = p[2];
    }
    __syncthreads();
    float bv = -1.f; int bi = 0x7fffffff;
    for (int n = t; n < N; n += 256) {
      const float* p = &xyz[((size_t)b * N + n) * 3];
      float dx = p[0] - cent[0], dy = p[1] - cent[1], dz = p[2] - cent[2];
      float d = dx * dx + dy * dy + dz * dz;
      float dd = fminf(dist[n], d);
      dist[n] = dd;
      if (dd > bv) { bv = dd; bi = n; }
    }
    rv[t] = bv; ri[t] = bi;
    __syncthreads();
    for (int st = 128; st > 0; st >>= 1) {
      if (t < st) {
        if (rv[t + st] > rv[t] || (rv[t + st] == rv[t] && ri[t + st] < ri[t])) {
          rv[t] = rv[t + st]; ri[t] = ri[t + st];
        }
      }
      __syncthreads();
    }
    if (t == 0) sfar = ri[0];
    __syncthreads();
    far = sfar;
  }
}

__global__ void gather_xyz_kernel(const float* __restrict__ xyz,
                                  const int* __restrict__ idx, int S, int N,
                                  float* __restrict__ out) {
  int i = blockIdx.x * 256 + threadIdx.x;
  if (i >= BATCH * S) return;
  int b = i / S;
  int id = idx[i];
  const float* p = &xyz[((size_t)b * N + id) * 3];
  float* o = &out[(size_t)i * 3];
  o[0] = p[0]; o[1] = p[1]; o[2] = p[2];
}

// kNN (32 smallest) via per-thread insertion list.
__global__ void knn_kernel(const float* __restrict__ q,   // (B,S,3)
                           const float* __restrict__ pts, // (B,N,3)
                           int S, int N, int* __restrict__ out) {
  int i = blockIdx.x * 256 + threadIdx.x;
  if (i >= BATCH * S) return;
  int b = i / S;
  const float* qp = &q[(size_t)i * 3];
  float qx = qp[0], qy = qp[1], qz = qp[2];
  float bd[32]; int bi[32];
  for (int j = 0; j < 32; ++j) { bd[j] = 3.4e38f; bi[j] = 0; }
  const float* pb = &pts[(size_t)b * N * 3];
  for (int n = 0; n < N; ++n) {
    float dx = pb[n * 3 + 0] - qx, dy = pb[n * 3 + 1] - qy, dz = pb[n * 3 + 2] - qz;
    float d = dx * dx + dy * dy + dz * dz;
    if (d < bd[31]) {
      int p = 31;
      while (p > 0 && bd[p - 1] > d) { bd[p] = bd[p - 1]; bi[p] = bi[p - 1]; --p; }
      bd[p] = d; bi[p] = n;
    }
  }
  for (int j = 0; j < 32; ++j) out[(size_t)i * 32 + j] = bi[j];
}

// max over k=32 neighbors: (B,C,S*32) -> (B,C,S) into a (DCB,doff) slot
__global__ void maxpool_k_kernel(const _Float16* __restrict__ src, int C, int S,
                                 _Float16* __restrict__ dst, int DCB, int doff) {
  int i = blockIdx.x * 256 + threadIdx.x;
  if (i >= BATCH * C * S) return;
  int s = i % S, c = (i / S) % C, b = i / (S * C);
  const _Float16* p = src + ((size_t)b * C + c) * (size_t)(S * 32) + s * 32;
  float m = -3.4e38f;
  for (int k = 0; k < 32; ++k) m = fmaxf(m, (float)p[k]);
  dst[((size_t)b * DCB + doff + c) * S + s] = (_Float16)m;
}

__global__ void softmax_row_kernel(float* __restrict__ attn) {  // rows of 256
  int row = blockIdx.x, t = threadIdx.x;
  float* p = attn + (size_t)row * 256;
  __shared__ float red[256];
  float v = p[t];
  red[t] = v; __syncthreads();
  for (int st = 128; st > 0; st >>= 1) {
    if (t < st) red[t] = fmaxf(red[t], red[t + st]);
    __syncthreads();
  }
  float mx = red[0]; __syncthreads();
  float e = __expf(v - mx);
  red[t] = e; __syncthreads();
  for (int st = 128; st > 0; st >>= 1) {
    if (t < st) red[t] += red[t + st];
    __syncthreads();
  }
  p[t] = e / red[0];
}

__global__ void colnorm_kernel(float* __restrict__ attn) {  // / (1e-9 + colsum)
  int b = blockIdx.x, n2 = threadIdx.x;
  float* base = attn + (size_t)b * 256 * 256;
  float s = 0.f;
  for (int n1 = 0; n1 < 256; ++n1) s += base[n1 * 256 + n2];
  float inv = 1.f / (1e-9f + s);
  for (int n1 = 0; n1 < 256; ++n1) base[n1 * 256 + n2] *= inv;
}

__global__ void diff_kernel(const _Float16* __restrict__ h, int HCB, int hoff,
                            const _Float16* __restrict__ xr,
                            _Float16* __restrict__ out, int C, int M, int total) {
  int i = blockIdx.x * 256 + threadIdx.x;
  if (i >= total) return;
  int m = i % M, c = (i / M) % C, b = i / (M * C);
  out[((size_t)b * C + c) * M + m] =
      (_Float16)((float)h[((size_t)b * HCB + hoff + c) * M + m] -
                 (float)xr[((size_t)b * C + c) * M + m]);
}

__global__ void finalmax_kernel(const _Float16* __restrict__ fz,
                                float* __restrict__ out, int C, int Mn) {
  int i = blockIdx.x * 256 + threadIdx.x;
  if (i >= BATCH * C) return;
  const _Float16* p = fz + (size_t)i * Mn;
  float m = -3.4e38f;
  for (int n = 0; n < Mn; ++n) m = fmaxf(m, (float)p[n]);
  out[i] = m;
}

// ----------------------------- host helpers --------------------------------

static void run_gemm_h(const _Float16* Apk, size_t aPB,
                       const _Float16* X, int XCB, int Xoff,
                       _Float16* Y, int YOB, int Yoff, const float* bias,
                       int M, int C, int O, hipStream_t s) {
  dim3 g(M / 64, (O + 31) / 32, BATCH);
  gemm16_kernel<_Float16><<<g, 256, 0, s>>>(Apk, aPB, X, XCB, Xoff, Y, YOB,
                                            Yoff, bias, M, C, O);
}
static void run_gemm_f(const _Float16* Apk, size_t aPB,
                       const _Float16* X, int XCB, int Xoff,
                       float* Y, int YOB, int Yoff, const float* bias,
                       int M, int C, int O, hipStream_t s) {
  dim3 g(M / 64, (O + 31) / 32, BATCH);
  gemm16_kernel<float><<<g, 256, 0, s>>>(Apk, aPB, X, XCB, Xoff, Y, YOB,
                                         Yoff, bias, M, C, O);
}
static void run_bn(const _Float16* src, int SCB, int soff,
                   _Float16* dst, int DCB, int doff,
                   const _Float16* res, int RCB, int roff,
                   const float* g, const float* bt, float* stats,
                   int C, int M, int mode, hipStream_t s) {
  float* sums = stats;
  float* sumsq = stats + 1024;
  zero_kernel<<<(2048 + 255) / 256, 256, 0, s>>>(stats, 2048);
  bn_stats_kernel<<<dim3(C, BATCH), 256, 0, s>>>(src, SCB, soff, M, sums, sumsq);
  int total = BATCH * C * M;
  bn_apply_kernel<<<(total + 255) / 256, 256, 0, s>>>(
      src, SCB, soff, dst, DCB, doff, res, RCB, roff, g, bt, sums, sumsq,
      1.0f / ((float)BATCH * (float)M), C, M, total, mode);
}

// ---------------------------------------------------------------------------
extern "C" void kernel_launch(void* const* d_in, const int* in_sizes, int n_in,
                              void* d_out, int out_size, void* d_ws, size_t ws_size,
                              hipStream_t stream) {
  (void)in_sizes; (void)n_in; (void)out_size; (void)ws_size;
  // ---- inputs (JAX pytree order: dict keys sorted, DFS) ----
  const float* att_b1   = (const float*)d_in[0];
  const float* att_b2   = (const float*)d_in[1];
  const float* att_c1w  = (const float*)d_in[2];
  const float* att_c2w  = (const float*)d_in[3];
  const float* att_g1   = (const float*)d_in[4];
  const float* att_g2   = (const float*)d_in[5];
  const float *sa_b[4], *sa_g[4], *sa_qkw[4], *sa_tb[4], *sa_tw[4], *sa_vb[4], *sa_vw[4];
  for (int i = 0; i < 4; ++i) {
    sa_b[i]   = (const float*)d_in[6 + 7 * i + 0];
    sa_g[i]   = (const float*)d_in[6 + 7 * i + 1];
    sa_qkw[i] = (const float*)d_in[6 + 7 * i + 2];
    sa_tb[i]  = (const float*)d_in[6 + 7 * i + 3];
    sa_tw[i]  = (const float*)d_in[6 + 7 * i + 4];
    sa_vb[i]  = (const float*)d_in[6 + 7 * i + 5];
    sa_vw[i]  = (const float*)d_in[6 + 7 * i + 6];
  }
  const float* bn1_b   = (const float*)d_in[34];
  const float* bn1_g   = (const float*)d_in[35];
  const float* bn2_b   = (const float*)d_in[36];
  const float* bn2_g   = (const float*)d_in[37];
  const float* conv1_w = (const float*)d_in[38];
  const float* conv2_w = (const float*)d_in[39];
  const float* fuse_b  = (const float*)d_in[40];
  const float* fuse_g  = (const float*)d_in[41];
  const float* fuse_w  = (const float*)d_in[42];
  const float* lo0_b1  = (const float*)d_in[43];
  const float* lo0_b2  = (const float*)d_in[44];
  const float* lo0_g1  = (const float*)d_in[45];
  const float* lo0_g2  = (const float*)d_in[46];
  const float* lo0_w1  = (const float*)d_in[47];
  const float* lo0_w2  = (const float*)d_in[48];
  const float* lo1_b1  = (const float*)d_in[49];
  const float* lo1_b2  = (const float*)d_in[50];
  const float* lo1_g1  = (const float*)d_in[51];
  const float* lo1_g2  = (const float*)d_in[52];
  const float* lo1_w1  = (const float*)d_in[53];
  const float* lo1_w2  = (const float*)d_in[54];
  const float* x       = (const float*)d_in[55];

  // ---- workspace carve-up ----
  char* cur = (char*)d_ws;
  auto take = [&](size_t bytes) -> void* {
    void* p = (void*)cur;
    cur += (bytes + 255) & ~(size_t)255;
    return p;
  };
  float* stats = (float*)take(2048 * sizeof(float));
  auto takeH = [&](size_t n) { return (_Float16*)take(n * sizeof(_Float16)); };
  _Float16* wpk_conv2 = takeH(64 * 64);
  _Float16* wpk_lo0_1 = takeH(128 * 128);
  _Float16* wpk_lo0_2 = takeH(128 * 128);
  _Float16* wpk_lo1_1 = takeH(256 * 256);
  _Float16* wpk_lo1_2 = takeH(256 * 256);
  _Float16* wpk_att1  = takeH(256 * 256);
  _Float16* wpk_att2  = takeH(256 * 256);
  _Float16 *wpk_qk[4], *wpk_v[4], *wpk_t[4];
  for (int i = 0; i < 4; ++i) {
    wpk_qk[i] = takeH(64 * 256);
    wpk_v[i]  = takeH(256 * 256);
    wpk_t[i]  = takeH(256 * 256);
  }
  _Float16* wpk_fuse = takeH(1024 * 1280);
  int*   fps1  = (int*)take((size_t)BATCH * 512 * 4);
  int*   fps2  = (int*)take((size_t)BATCH * 256 * 4);
  int*   knn1  = (int*)take((size_t)BATCH * 512 * 32 * 4);
  int*   knn2  = (int*)take((size_t)BATCH * 256 * 32 * 4);
  float* nxyz1 = (float*)take((size_t)BATCH * 512 * 3 * 4);
  float* nxyz2 = (float*)take((size_t)BATCH * 256 * 3 * 4);
  _Float16* f0     = takeH((size_t)BATCH * 128 * 512);
  _Float16* bigcat = takeH((size_t)BATCH * 1280 * 256);  // sa0..3 | f1
  _Float16* proj   = takeH((size_t)BATCH * 64 * 256);
  _Float16* projpk = takeH((size_t)BATCH * 256 * 64);    // packed Gram A
  _Float16* vbuf   = takeH((size_t)BATCH * 256 * 256);
  _Float16* vpk    = takeH((size_t)BATCH * 256 * 256);   // packed x_r A
  float*    attnF  = (float*)take((size_t)BATCH * 256 * 256 * 4);
  _Float16* attn16 = takeH((size_t)BATCH * 256 * 256);
  _Float16* xrbuf  = takeH((size_t)BATCH * 256 * 256);
  _Float16* dbuf   = takeH((size_t)BATCH * 256 * 256);
  _Float16* tbuf   = takeH((size_t)BATCH * 256 * 256);
  _Float16* fz     = takeH((size_t)BATCH * 1024 * 256);
  _Float16* bufA   = takeH((size_t)BATCH * 128 * 16384);  // 128 MiB ping
  _Float16* bufB   = takeH((size_t)BATCH * 128 * 16384);  // 128 MiB pong

  // ---- weights -> packed f16 WMMA layout (cvt fused into pack) ----
#define PACKW(SRC, DST, O_, C_)                                          \
  packA_kernel<float><<<((O_) * (C_) + 255) / 256, 256, 0, stream>>>(    \
      SRC, 0, C_, 1, DST, O_, C_, (O_) * (C_))
  PACKW(conv2_w, wpk_conv2, 64, 64);
  PACKW(lo0_w1, wpk_lo0_1, 128, 128);
  PACKW(lo0_w2, wpk_lo0_2, 128, 128);
  PACKW(lo1_w1, wpk_lo1_1, 256, 256);
  PACKW(lo1_w2, wpk_lo1_2, 256, 256);
  PACKW(att_c1w, wpk_att1, 256, 256);
  PACKW(att_c2w, wpk_att2, 256, 256);
  for (int i = 0; i < 4; ++i) {
    PACKW(sa_qkw[i], wpk_qk[i], 64, 256);
    PACKW(sa_vw[i], wpk_v[i], 256, 256);
    PACKW(sa_tw[i], wpk_t[i], 256, 256);
  }
  PACKW(fuse_w, wpk_fuse, 1024, 1280);
#undef PACKW

  // ---- conv1 + BN + relu ; conv2 + BN + relu  (h2 = bufB, (B,64,2048)) ----
  {
    int tot = BATCH * 64 * 2048;
    conv1_kernel<<<(tot + 255) / 256, 256, 0, stream>>>(x, conv1_w, bufA);
    run_bn(bufA, 64, 0, bufA, 64, 0, nullptr, 0, 0, bn1_g, bn1_b, stats, 64, 2048, 0, stream);
    run_gemm_h(wpk_conv2, 0, bufA, 64, 0, bufB, 64, 0, nullptr, 2048, 64, 64, stream);
    run_bn(bufB, 64, 0, bufB, 64, 0, nullptr, 0, 0, bn2_g, bn2_b, stats, 64, 2048, 0, stream);
  }

  // ---- sample_and_group(512,32) + local_op(128) -> f0 (B,128,512) ----
  fps_kernel<<<BATCH, 256, 0, stream>>>(x, 2048, 512, fps1);
  gather_xyz_kernel<<<(BATCH * 512 + 255) / 256, 256, 0, stream>>>(x, fps1, 512, 2048, nxyz1);
  knn_kernel<<<(BATCH * 512 + 255) / 256, 256, 0, stream>>>(nxyz1, x, 512, 2048, knn1);
  {
    dim3 g(16384 / 64, 128 / 32, BATCH);
    gemm16_gather_kernel<<<g, 256, 0, stream>>>(wpk_lo0_1, bufB, 64, 2048, knn1,
                                                fps1, 512, bufA, 16384, 128, 128);
  }
  run_bn(bufA, 128, 0, bufA, 128, 0, nullptr, 0, 0, lo0_g1, lo0_b1, stats, 128, 16384, 0, stream);
  run_gemm_h(wpk_lo0_2, 0, bufA, 128, 0, bufB, 128, 0, nullptr, 16384, 128, 128, stream);
  run_bn(bufB, 128, 0, bufB, 128, 0, nullptr, 0, 0, lo0_g2, lo0_b2, stats, 128, 16384, 0, stream);
  maxpool_k_kernel<<<(BATCH * 128 * 512 + 255) / 256, 256, 0, stream>>>(bufB, 128, 512, f0, 128, 0);

  // ---- sample_and_group(256,32) + local_op(256) -> f1 (bigcat ch 1024..1279) ----
  fps_kernel<<<BATCH, 256, 0, stream>>>(nxyz1, 512, 256, fps2);
  gather_xyz_kernel<<<(BATCH * 256 + 255) / 256, 256, 0, stream>>>(nxyz1, fps2, 256, 512, nxyz2);
  knn_kernel<<<(BATCH * 256 + 255) / 256, 256, 0, stream>>>(nxyz2, nxyz1, 256, 512, knn2);
  {
    dim3 g(8192 / 64, 256 / 32, BATCH);
    gemm16_gather_kernel<<<g, 256, 0, stream>>>(wpk_lo1_1, f0, 128, 512, knn2,
                                                fps2, 256, bufA, 8192, 256, 256);
  }
  run_bn(bufA, 256, 0, bufA, 256, 0, nullptr, 0, 0, lo1_g1, lo1_b1, stats, 256, 8192, 0, stream);
  run_gemm_h(wpk_lo1_2, 0, bufA, 256, 0, bufB, 256, 0, nullptr, 8192, 256, 256, stream);
  run_bn(bufB, 256, 0, bufB, 256, 0, nullptr, 0, 0, lo1_g2, lo1_b2, stats, 256, 8192, 0, stream);
  maxpool_k_kernel<<<(BATCH * 256 * 256 + 255) / 256, 256, 0, stream>>>(bufB, 256, 256, bigcat, 1280, 1024);

  // ---- attention stem: conv1/conv2 on f1 ----
  run_gemm_h(wpk_att1, 0, bigcat, 1280, 1024, bufA, 256, 0, nullptr, 256, 256, 256, stream);
  run_bn(bufA, 256, 0, bufA, 256, 0, nullptr, 0, 0, att_g1, att_b1, stats, 256, 256, 0, stream);
  run_gemm_h(wpk_att2, 0, bufA, 256, 0, bufB, 256, 0, nullptr, 256, 256, 256, stream);
  run_bn(bufB, 256, 0, bufB, 256, 0, nullptr, 0, 0, att_g2, att_b2, stats, 256, 256, 0, stream);

  // ---- 4 SA layers; output i -> bigcat channels [i*256, (i+1)*256) ----
  for (int i = 0; i < 4; ++i) {
    const _Float16* h; int HCB, hoff;
    if (i == 0) { h = bufB; HCB = 256; hoff = 0; }
    else        { h = bigcat; HCB = 1280; hoff = (i - 1) * 256; }
    // proj = qk_w @ h  (shared q/k projection)
    run_gemm_h(wpk_qk[i], 0, h, HCB, hoff, proj, 64, 0, nullptr, 256, 256, 64, stream);
    // v = v_w @ h + v_b
    run_gemm_h(wpk_v[i], 0, h, HCB, hoff, vbuf, 256, 0, sa_vb[i], 256, 256, 256, stream);
    // attn_pre[n1][n2] = sum_d proj[d][n1] * proj[d][n2]  (Gram; A = proj^T, packed)
    packA_kernel<_Float16><<<(BATCH * 256 * 64 + 255) / 256, 256, 0, stream>>>(
        proj, (size_t)64 * 256, 1, 256, projpk, 256, 64, BATCH * 256 * 64);
    run_gemm_f(projpk, (size_t)256 * 64, proj, 64, 0, attnF, 256, 0, nullptr, 256, 64, 256, stream);
    softmax_row_kernel<<<BATCH * 256, 256, 0, stream>>>(attnF);
    colnorm_kernel<<<BATCH, 256, 0, stream>>>(attnF);
    cvt_f32_f16_kernel<<<(BATCH * 256 * 256 + 255) / 256, 256, 0, stream>>>(attnF, attn16, BATCH * 256 * 256);
    // x_r = v @ attn  (per-batch A matrix, packed)
    packA_kernel<_Float16><<<(BATCH * 256 * 256 + 255) / 256, 256, 0, stream>>>(
        vbuf, (size_t)256 * 256, 256, 1, vpk, 256, 256, BATCH * 256 * 256);
    run_gemm_h(vpk, (size_t)256 * 256, attn16, 256, 0, xrbuf, 256, 0, nullptr, 256, 256, 256, stream);
    // t = t_w @ (h - x_r) + t_b ; out = h + relu(bn(t))
    diff_kernel<<<(BATCH * 256 * 256 + 255) / 256, 256, 0, stream>>>(h, HCB, hoff, xrbuf, dbuf, 256, 256, BATCH * 256 * 256);
    run_gemm_h(wpk_t[i], 0, dbuf, 256, 0, tbuf, 256, 0, sa_tb[i], 256, 256, 256, stream);
    run_bn(tbuf, 256, 0, bigcat, 1280, i * 256, h, HCB, hoff, sa_g[i], sa_b[i], stats, 256, 256, 0, stream);
  }

  // ---- fuse: (1024 x 1280) over concat, BN + leaky(0.2), max over n ----
  run_gemm_h(wpk_fuse, 0, bigcat, 1280, 0, fz, 1024, 0, nullptr, 256, 1280, 1024, stream);
  run_bn(fz, 1024, 0, fz, 1024, 0, nullptr, 0, 0, fuse_g, fuse_b, stats, 1024, 256, 1, stream);
  finalmax_kernel<<<(BATCH * 1024 + 255) / 256, 256, 0, stream>>>(fz, (float*)d_out, 1024, 256);
}
